// GNN_15144054685741
// MI455X (gfx1250) — matrix operand order repaired and checked by reading, not compiled, not run on hardware
//
#include <hip/hip_runtime.h>
#include <math.h>

#define DIN 128
#define HID 256
#define NCL 10
#define KSEL 100
#define KMIT 10

typedef __attribute__((ext_vector_type(16))) __bf16 v16bf;
typedef __attribute__((ext_vector_type(8)))  float  v8f;

union Frag { v16bf v; unsigned u[8]; };

__device__ __forceinline__ unsigned short f2bf(float f) {
  unsigned u = __float_as_uint(f);
  u += 0x7FFFu + ((u >> 16) & 1u);            // round-to-nearest-even
  return (unsigned short)(u >> 16);
}
__device__ __forceinline__ unsigned packbf2(float lo, float hi) {
  return (unsigned)f2bf(lo) | ((unsigned)f2bf(hi) << 16);
}

// ---------------- utility kernels ----------------
__global__ void fill_f32(float* p, float v, int n) {
  int i = blockIdx.x * blockDim.x + threadIdx.x;
  if (i < n) p[i] = v;
}
__global__ void fill_i32(int* p, int v, int n) {
  int i = blockIdx.x * blockDim.x + threadIdx.x;
  if (i < n) p[i] = v;
}
// row-major f32 -> packed bf16 pairs (dword i = {X[2i], X[2i+1]})
__global__ void pack_pairs_bf16(const float* __restrict__ X, unsigned* __restrict__ Xb, int nPairs) {
  int i = blockIdx.x * blockDim.x + threadIdx.x;
  if (i >= nPairs) return;
  const float2 x = *(const float2*)(X + 2 * (size_t)i);
  Xb[i] = packbf2(x.x, x.y);
}
// same, with fused relu (conv1 output -> conv2 GEMM input)
__global__ void relu_pack_pairs_bf16(const float* __restrict__ X, unsigned* __restrict__ Xb, int nPairs) {
  int i = blockIdx.x * blockDim.x + threadIdx.x;
  if (i >= nPairs) return;
  const float2 x = *(const float2*)(X + 2 * (size_t)i);
  Xb[i] = packbf2(fmaxf(x.x, 0.0f), fmaxf(x.y, 0.0f));
}

// Pack fp32 weights [K x N] into FRAGMENT-MAJOR bf16 pairs:
// fragment tile (s = k0/32, t = n0/16) occupies 256 consecutive dwords;
// lane l (g=l>>4, r=l&15) owns dwords [l*8, l*8+8), dword v = {W[k][n], W[k+1][n]}
// with k = 32s + 16(v>>2) + 8g + 2(v&3), n = 16t + r.  => B fragment = two b128 loads per lane.
__global__ void pack_weights_frag(const float* __restrict__ W, unsigned* __restrict__ Wf,
                                  int K, int N) {
  int i = blockIdx.x * blockDim.x + threadIdx.x;
  int tot = (K >> 1) * N;
  if (i >= tot) return;
  int v = i & 7;
  int l = (i >> 3) & 31;
  int ft = i >> 8;                  // s * (N/16) + t
  int t = ft % (N >> 4);
  int s = ft / (N >> 4);
  int g = l >> 4, r = l & 15;
  int k = (s << 5) + ((v >> 2) << 4) + (g << 3) + ((v & 3) << 1);
  int n = (t << 4) + r;
  Wf[i] = packbf2(W[(size_t)k * N + n], W[(size_t)(k + 1) * N + n]);
}

// ---------------- degree / normalization ----------------
__global__ void deg_edges(const int* __restrict__ dst, const float* __restrict__ ew,
                          float* __restrict__ deg, int E) {
  int e = blockIdx.x * blockDim.x + threadIdx.x;
  if (e < E) atomicAdd(&deg[dst[e]], ew[e]);
}
__global__ void rsqrt_inplace(float* p, int n) {
  int i = blockIdx.x * blockDim.x + threadIdx.x;
  if (i < n) p[i] = rsqrtf(fmaxf(p[i], 1e-12f));
}

// A fragment from packed-bf16 row (dword pointer already at this row's k0/2):
// u[0..3] at +4g, u[4..7] at +8+4g (per 16-bit A-matrix lane layout).
__device__ __forceinline__ void load_afrag_bf(Frag& a, const unsigned* __restrict__ rowk0d, int g) {
  const unsigned* ap = rowk0d + 4 * g;
  *(uint4*)&a.u[0] = *(const uint4*)(ap + 0);
  *(uint4*)&a.u[4] = *(const uint4*)(ap + 8);
}
__device__ __forceinline__ void load_bfrag(Frag& b, const unsigned* __restrict__ Wf,
                                           int fragTile, int lane) {
  const unsigned* bp = Wf + (((size_t)fragTile << 5) + lane) * 8;
  *(uint4*)&b.u[0] = *(const uint4*)(bp + 0);
  *(uint4*)&b.u[4] = *(const uint4*)(bp + 4);
}

// ---------------- WMMA GEMM: Y[M,N] = X[M,K] @ W[K,N] ----------------
// Xb = packed bf16 pairs, row stride K/2 dwords; Wf = fragment-major bf16.
template <int K, int N>
__global__ __launch_bounds__(256) void wmma_gemm(const unsigned* __restrict__ Xb,
                                                 const unsigned* __restrict__ Wf,
                                                 float* __restrict__ Y, int M) {
  const int lane = threadIdx.x & 31, wid = threadIdx.x >> 5;
  const int g = lane >> 4, r = lane & 15;
  const int mt = blockIdx.x * 8 + wid;
  if (mt >= (M >> 4)) return;                  // wave-uniform: EXEC all-ones inside
  const int col0 = blockIdx.y * 64;
  const int row = mt * 16 + r;
  constexpr int NT16 = N / 16;
  v8f acc[4];
#pragma unroll
  for (int t = 0; t < 4; ++t) acc[t] = (v8f){};
#pragma unroll
  for (int k0 = 0; k0 < K; k0 += 32) {
    const int s = k0 >> 5;
    Frag a;
    load_afrag_bf(a, Xb + (size_t)row * (K / 2) + (k0 >> 1), g);
    Frag b[4];
#pragma unroll
    for (int nt = 0; nt < 4; ++nt)
      load_bfrag(b[nt], Wf, s * NT16 + (blockIdx.y * 4 + nt), lane);
#pragma unroll
    for (int nt = 0; nt < 4; ++nt)
      acc[nt] = __builtin_amdgcn_wmma_f32_16x16x32_bf16(false, a.v, false, b[nt].v,
                                                        (short)0, acc[nt], false, false);
  }
#pragma unroll
  for (int nt = 0; nt < 4; ++nt) {
    int col = col0 + nt * 16 + r;
#pragma unroll
    for (int v = 0; v < 8; ++v) {
      int orow = mt * 16 + (g << 3) + v;
      Y[(size_t)orow * N + col] = acc[nt][v];
    }
  }
}

// ---------------- GCN aggregation ----------------
__global__ void agg_init(const float* __restrict__ H, const float* __restrict__ dinv,
                         const float* __restrict__ bias, float* __restrict__ OUT,
                         int N, int F, int stride, int off) {
  int i = blockIdx.x * blockDim.x + threadIdx.x;
  if (i >= N * F) return;
  int n = i / F, f = i % F;
  float di = dinv[n];
  OUT[(size_t)n * stride + off + f] = H[(size_t)n * F + f] * di * di + bias[f];
}
__global__ void agg_edges(const int* __restrict__ src, const int* __restrict__ dst,
                          const float* __restrict__ ew, const float* __restrict__ dinv,
                          const float* __restrict__ H, float* __restrict__ OUT,
                          int E, int F, int stride, int off) {
  int F4 = F >> 2;
  int i = blockIdx.x * blockDim.x + threadIdx.x;
  if (i >= E * F4) return;
  int e = i / F4, f = (i % F4) << 2;
  int s = src[e], d = dst[e];
  float norm = dinv[s] * ew[e] * dinv[d];
  const float4 h = *(const float4*)(H + (size_t)s * F + f);
  float* o = OUT + (size_t)d * stride + off + f;
  atomicAdd(o + 0, h.x * norm);
  atomicAdd(o + 1, h.y * norm);
  atomicAdd(o + 2, h.z * norm);
  atomicAdd(o + 3, h.w * norm);
}

// ---------------- kmeans ----------------
__global__ void build_nodeof(const int* __restrict__ atk, int A, int* __restrict__ node_of, int N) {
  int i = blockIdx.x * blockDim.x + threadIdx.x;
  if (i >= N) return;
  int lo = 0, hi = A;
  while (lo < hi) { int mid = (lo + hi) >> 1; if (atk[mid] < i) lo = mid + 1; else hi = mid; }
  bool attacked = (lo < A) && (atk[lo] == i);
  if (!attacked) node_of[i - lo] = i;
}
__global__ void compute_xx(const float* __restrict__ emb, const int* __restrict__ node_of,
                           float* __restrict__ xx, int M) {
  int m = blockIdx.x * blockDim.x + threadIdx.x;
  if (m >= M) return;
  const float* row = emb + (size_t)node_of[m] * (2 * DIN);
  float s = 0.f;
  for (int f = 0; f < 2 * DIN; ++f) { float v = row[f]; s += v * v; }
  xx[m] = s;
}
__global__ void init_cent(const float* __restrict__ emb, const int* __restrict__ node_of,
                          float* __restrict__ cent) {
  int i = blockIdx.x * blockDim.x + threadIdx.x;
  if (i >= NCL * 2 * DIN) return;
  int c = i / (2 * DIN), f = i % (2 * DIN);
  cent[i] = emb[(size_t)node_of[c] * (2 * DIN) + f];
}
__global__ __launch_bounds__(256) void kmeans_assign(const float* __restrict__ emb,
    const int* __restrict__ node_of, const float* __restrict__ xx,
    const float* __restrict__ cent, float* __restrict__ csum, float* __restrict__ ccnt,
    int* __restrict__ assign, unsigned* __restrict__ key, int M, int final_pass) {
  __shared__ float sct[2 * DIN * NCL];   // transposed: [f][c]
  __shared__ float scc[NCL];
  for (int i = threadIdx.x; i < NCL * 2 * DIN; i += blockDim.x) {
    int c = i / (2 * DIN), f = i % (2 * DIN);
    sct[f * NCL + c] = cent[i];
  }
  __syncthreads();
  if (threadIdx.x < NCL) {
    float s = 0.f;
    for (int f = 0; f < 2 * DIN; ++f) { float v = sct[f * NCL + threadIdx.x]; s += v * v; }
    scc[threadIdx.x] = s;
  }
  __syncthreads();
  for (int m = blockIdx.x * blockDim.x + threadIdx.x; m < M; m += gridDim.x * blockDim.x) {
    const float* row = emb + (size_t)node_of[m] * (2 * DIN);
    float dot[NCL];
#pragma unroll
    for (int c = 0; c < NCL; ++c) dot[c] = 0.f;
    for (int f = 0; f < 2 * DIN; ++f) {
      float x = row[f];
#pragma unroll
      for (int c = 0; c < NCL; ++c) dot[c] += x * sct[f * NCL + c];
    }
    int best = 0;
    float bestd = xx[m] - 2.f * dot[0] + scc[0];
#pragma unroll
    for (int c = 1; c < NCL; ++c) {
      float d = xx[m] - 2.f * dot[c] + scc[c];
      if (d < bestd) { bestd = d; best = c; }
    }
    if (final_pass) {
      assign[m] = best;
      unsigned u = __float_as_uint(bestd);
      u = (u & 0x80000000u) ? ~u : (u | 0x80000000u);   // order-preserving map
      key[m] = u;
    } else {
      for (int f = 0; f < 2 * DIN; ++f) atomicAdd(&csum[best * (2 * DIN) + f], row[f]);
      atomicAdd(&ccnt[best], 1.0f);
    }
  }
}
__global__ void update_cent(const float* __restrict__ csum, const float* __restrict__ ccnt,
                            float* __restrict__ cent) {
  int i = blockIdx.x * blockDim.x + threadIdx.x;
  if (i >= NCL * 2 * DIN) return;
  cent[i] = csum[i] / fmaxf(ccnt[i / (2 * DIN)], 1.0f);
}
// one block per cluster: binary search for the KSEL-th smallest key
__global__ __launch_bounds__(256) void topk_thresh(const unsigned* __restrict__ key,
    const int* __restrict__ assign, int M, unsigned* __restrict__ kth, int* __restrict__ needEq) {
  int c = blockIdx.x;
  __shared__ int red[256];
  unsigned lo = 0u, hi = 0xFFFFFFFFu;
  for (int it = 0; it < 32; ++it) {
    unsigned mid = lo + ((hi - lo) >> 1);
    int cnt = 0;
    for (int m = threadIdx.x; m < M; m += 256)
      if (assign[m] == c && key[m] <= mid) cnt++;
    red[threadIdx.x] = cnt;
    __syncthreads();
    for (int s = 128; s > 0; s >>= 1) {
      if (threadIdx.x < s) red[threadIdx.x] += red[threadIdx.x + s];
      __syncthreads();
    }
    int total = red[0];
    __syncthreads();
    if (total >= KSEL) hi = mid; else lo = mid + 1;
  }
  int less = 0;
  for (int m = threadIdx.x; m < M; m += 256)
    if (assign[m] == c && key[m] < lo) less++;
  red[threadIdx.x] = less;
  __syncthreads();
  for (int s = 128; s > 0; s >>= 1) {
    if (threadIdx.x < s) red[threadIdx.x] += red[threadIdx.x + s];
    __syncthreads();
  }
  if (threadIdx.x == 0) { kth[c] = lo; needEq[c] = KSEL - red[0]; }
}
__global__ void topk_select(const unsigned* __restrict__ key, const int* __restrict__ assign,
                            const int* __restrict__ node_of, const unsigned* __restrict__ kth,
                            const int* __restrict__ needEq, int* __restrict__ ctr,
                            float* __restrict__ umask, int M) {
  int m = blockIdx.x * blockDim.x + threadIdx.x;
  if (m >= M) return;
  int c = assign[m];
  unsigned k = key[m];
  if (k < kth[c]) umask[node_of[m]] = 1.0f;
  else if (k == kth[c]) {
    if (atomicAdd(&ctr[c], 1) < needEq[c]) umask[node_of[m]] = 1.0f;
  }
}

// ---------------- fused edge MLP (WMMA bf16) ----------------
// per 16-edge tile: gather bf16 [u_emb[src] | i_x_a[dst]] (K=384), 384->128 relu,
// 128->32 relu, 32->1 sigmoid * mask
__global__ __launch_bounds__(256) void mlp_edges(const int* __restrict__ src,
    const int* __restrict__ dst, const unsigned* __restrict__ uembb,
    const unsigned* __restrict__ ixab,
    const unsigned* __restrict__ W1f, const float* __restrict__ b1,
    const unsigned* __restrict__ W2f, const float* __restrict__ b2,
    const float* __restrict__ W3, const float* __restrict__ b3,
    const float* __restrict__ umask, float* __restrict__ out, int E) {
  __shared__ unsigned short h1s[8][16][128];   // per-wave bf16 h1 (wave-private, DS in-order)
  __shared__ float h2s[8][16][32];
  const int lane = threadIdx.x & 31, wid = threadIdx.x >> 5;
  const int g = lane >> 4, r = lane & 15;
  const int tile = blockIdx.x * 8 + wid;
  if (tile >= (E >> 4)) return;                // wave-uniform
  const int e_r = tile * 16 + r;
  const int s_node = src[e_r], d_node = dst[e_r];
  const unsigned* urow = uembb + (size_t)s_node * DIN;       // 256 bf16 = 128 dwords
  const unsigned* irow = ixab + (size_t)d_node * (DIN / 2);  // 128 bf16 = 64 dwords

  // ---- layer 1: [16 x 384] @ [384 x 128] ----
  v8f acc[8];
#pragma unroll
  for (int t = 0; t < 8; ++t) acc[t] = (v8f){};
#pragma unroll 2
  for (int k0 = 0; k0 < 3 * DIN; k0 += 32) {
    // whole 32-wide k-step stays inside one source tensor
    const unsigned* rowk0d = (k0 < 2 * DIN) ? (urow + (k0 >> 1)) : (irow + ((k0 - 2 * DIN) >> 1));
    Frag a;
    load_afrag_bf(a, rowk0d, g);
    const int s = k0 >> 5;
    Frag b[8];
#pragma unroll
    for (int nt = 0; nt < 8; ++nt) load_bfrag(b[nt], W1f, s * 8 + nt, lane);
#pragma unroll
    for (int nt = 0; nt < 8; ++nt)
      acc[nt] = __builtin_amdgcn_wmma_f32_16x16x32_bf16(false, a.v, false, b[nt].v,
                                                        (short)0, acc[nt], false, false);
  }
  // bias + relu -> LDS (bf16), element (row = 8g+v, col = nt*16+r)
#pragma unroll
  for (int nt = 0; nt < 8; ++nt) {
    int col = nt * 16 + r;
#pragma unroll
    for (int v = 0; v < 8; ++v)
      h1s[wid][(g << 3) + v][col] = f2bf(fmaxf(acc[nt][v] + b1[col], 0.0f));
  }
  // ---- layer 2: [16 x 128] @ [128 x 32] ----
  v8f acc2[2];
#pragma unroll
  for (int t = 0; t < 2; ++t) acc2[t] = (v8f){};
  const uint4* hrow = (const uint4*)&h1s[wid][r][0];   // 16 uint4 per row
#pragma unroll
  for (int k0 = 0; k0 < 128; k0 += 32) {
    Frag a;
    const uint4 q0 = hrow[(k0 >> 3) + g];        // k = k0+8g .. +7
    const uint4 q1 = hrow[(k0 >> 3) + g + 2];    // k = k0+16+8g .. +7
    a.u[0] = q0.x; a.u[1] = q0.y; a.u[2] = q0.z; a.u[3] = q0.w;
    a.u[4] = q1.x; a.u[5] = q1.y; a.u[6] = q1.z; a.u[7] = q1.w;
    const int s = k0 >> 5;
    Frag b[2];
#pragma unroll
    for (int nt = 0; nt < 2; ++nt) load_bfrag(b[nt], W2f, s * 2 + nt, lane);
#pragma unroll
    for (int nt = 0; nt < 2; ++nt)
      acc2[nt] = __builtin_amdgcn_wmma_f32_16x16x32_bf16(false, a.v, false, b[nt].v,
                                                         (short)0, acc2[nt], false, false);
  }
#pragma unroll
  for (int nt = 0; nt < 2; ++nt) {
    int col = nt * 16 + r;
#pragma unroll
    for (int v = 0; v < 8; ++v)
      h2s[wid][(g << 3) + v][col] = fmaxf(acc2[nt][v] + b2[col], 0.0f);
  }
  // ---- layer 3: 32 -> 1, sigmoid, mask ----
  if (lane < 16) {
    float z = b3[0];
#pragma unroll
    for (int j = 0; j < 32; ++j) z += h2s[wid][lane][j] * W3[j];
    float sg = 1.0f / (1.0f + __expf(-z));
    out[tile * 16 + lane] = sg * umask[s_node];  // lane<16 => r==lane => s_node is this edge's src
  }
}

// ---------------- host ----------------
extern "C" void kernel_launch(void* const* d_in, const int* in_sizes, int n_in,
                              void* d_out, int out_size, void* d_ws, size_t ws_size,
                              hipStream_t stream) {
  const float* u_x  = (const float*)d_in[0];
  const float* i_x  = (const float*)d_in[1];
  const int*   eidx = (const int*)d_in[2];
  const float* ew1  = (const float*)d_in[3];
  const float* ew2  = (const float*)d_in[4];
  const int*   atk  = (const int*)d_in[5];
  const float* W1a = (const float*)d_in[6];  const float* b1a = (const float*)d_in[7];
  const float* W2a = (const float*)d_in[8];  const float* b2a = (const float*)d_in[9];
  const float* W1b = (const float*)d_in[10]; const float* b1b = (const float*)d_in[11];
  const float* W2b = (const float*)d_in[12]; const float* b2b = (const float*)d_in[13];
  const float* mW1 = (const float*)d_in[14]; const float* mb1 = (const float*)d_in[15];
  const float* mW2 = (const float*)d_in[16]; const float* mb2 = (const float*)d_in[17];
  const float* mW3 = (const float*)d_in[18]; const float* mb3 = (const float*)d_in[19];

  const int N = in_sizes[0] / DIN;
  const int E = in_sizes[3];
  const int A = in_sizes[5];
  const int M = N - A;
  const int* src = eidx;
  const int* dst = eidx + E;
  float* out = (float*)d_out;

  // workspace carve-out
  char* ws = (char*)d_ws;
  size_t off = 0;
  auto alloc = [&](size_t bytes) -> void* {
    void* p = ws + off;
    off += (bytes + 255) & ~(size_t)255;
    return p;
  };
  unsigned* W1a_f = (unsigned*)alloc((size_t)(DIN / 2) * HID * 4);
  unsigned* W2a_f = (unsigned*)alloc((size_t)(HID / 2) * DIN * 4);
  unsigned* W1b_f = (unsigned*)alloc((size_t)(DIN / 2) * HID * 4);
  unsigned* W2b_f = (unsigned*)alloc((size_t)(HID / 2) * DIN * 4);
  unsigned* mW1_f = (unsigned*)alloc((size_t)(3 * DIN / 2) * 128 * 4);
  unsigned* mW2_f = (unsigned*)alloc((size_t)(128 / 2) * 32 * 4);
  float* dinv1 = (float*)alloc((size_t)N * 4);
  float* dinv2 = (float*)alloc((size_t)N * 4);
  float* t0    = (float*)alloc((size_t)N * HID * 4);
  float* t1    = (float*)alloc((size_t)N * HID * 4);
  float* t2    = (float*)alloc((size_t)N * DIN * 4);
  float* uemb  = (float*)alloc((size_t)N * 2 * DIN * 4);
  float* ixa   = (float*)alloc((size_t)N * DIN * 4);
  unsigned* uxb  = (unsigned*)alloc((size_t)N * (DIN / 2) * 4);   // bf16-pair inputs
  unsigned* ixb  = (unsigned*)alloc((size_t)N * (DIN / 2) * 4);
  unsigned* t1b  = (unsigned*)alloc((size_t)N * (HID / 2) * 4);
  unsigned* uembb = (unsigned*)alloc((size_t)N * DIN * 4);        // 256 bf16 / row
  unsigned* ixab  = (unsigned*)alloc((size_t)N * (DIN / 2) * 4);
  int*   node_of = (int*)alloc((size_t)M * 4);
  float* xx    = (float*)alloc((size_t)M * 4);
  int*   assign = (int*)alloc((size_t)M * 4);
  unsigned* key = (unsigned*)alloc((size_t)M * 4);
  float* cent = (float*)alloc((size_t)NCL * 2 * DIN * 4);
  float* csum = (float*)alloc((size_t)NCL * 2 * DIN * 4);
  float* ccnt = (float*)alloc((size_t)NCL * 4);
  unsigned* kth = (unsigned*)alloc((size_t)NCL * 4);
  int* needEq = (int*)alloc((size_t)NCL * 4);
  int* ctr    = (int*)alloc((size_t)NCL * 4);
  float* umask = (float*)alloc((size_t)N * 4);
  (void)ws_size; (void)n_in; (void)out_size;

  const int BS = 256;
  auto cdiv = [](long a, long b) { return (int)((a + b - 1) / b); };

  // pack all weights to fragment-major bf16 pairs
  pack_weights_frag<<<cdiv((DIN / 2) * HID, BS), BS, 0, stream>>>(W1a, W1a_f, DIN, HID);
  pack_weights_frag<<<cdiv((HID / 2) * DIN, BS), BS, 0, stream>>>(W2a, W2a_f, HID, DIN);
  pack_weights_frag<<<cdiv((DIN / 2) * HID, BS), BS, 0, stream>>>(W1b, W1b_f, DIN, HID);
  pack_weights_frag<<<cdiv((HID / 2) * DIN, BS), BS, 0, stream>>>(W2b, W2b_f, HID, DIN);
  pack_weights_frag<<<cdiv((3 * DIN / 2) * 128, BS), BS, 0, stream>>>(mW1, mW1_f, 3 * DIN, 128);
  pack_weights_frag<<<cdiv((128 / 2) * 32, BS), BS, 0, stream>>>(mW2, mW2_f, 128, 32);

  // bf16 copies of the GEMM inputs (u_x reused by stacks a and b)
  pack_pairs_bf16<<<cdiv((long)N * (DIN / 2), BS), BS, 0, stream>>>(u_x, uxb, N * (DIN / 2));
  pack_pairs_bf16<<<cdiv((long)N * (DIN / 2), BS), BS, 0, stream>>>(i_x, ixb, N * (DIN / 2));

  // degree -> dinv for both edge-weight sets (self-loop weight 1.0 folded into init)
  fill_f32<<<cdiv(N, BS), BS, 0, stream>>>(dinv1, 1.0f, N);
  deg_edges<<<cdiv(E, BS), BS, 0, stream>>>(dst, ew1, dinv1, E);
  rsqrt_inplace<<<cdiv(N, BS), BS, 0, stream>>>(dinv1, N);
  fill_f32<<<cdiv(N, BS), BS, 0, stream>>>(dinv2, 1.0f, N);
  deg_edges<<<cdiv(E, BS), BS, 0, stream>>>(dst, ew2, dinv2, E);
  rsqrt_inplace<<<cdiv(N, BS), BS, 0, stream>>>(dinv2, N);

  const int gx = cdiv(N >> 4, 8);

  auto gcn_stack = [&](const unsigned* Xb, const unsigned* Wf1, const float* bias1,
                       const unsigned* Wf2, const float* bias2,
                       const float* ew, const float* dinv,
                       float* OUT, int ostride, int ooff) {
    // conv1: DIN -> HID, relu
    wmma_gemm<DIN, HID><<<dim3(gx, HID / 64), BS, 0, stream>>>(Xb, Wf1, t0, N);
    agg_init<<<cdiv((long)N * HID, BS), BS, 0, stream>>>(t0, dinv, bias1, t1, N, HID, HID, 0);
    agg_edges<<<cdiv((long)E * (HID / 4), BS), BS, 0, stream>>>(src, dst, ew, dinv, t0, t1, E, HID, HID, 0);
    // fused relu + bf16 pack of conv1 output
    relu_pack_pairs_bf16<<<cdiv((long)N * (HID / 2), BS), BS, 0, stream>>>(t1, t1b, N * (HID / 2));
    // conv2: HID -> DIN
    wmma_gemm<HID, DIN><<<dim3(gx, DIN / 64), BS, 0, stream>>>(t1b, Wf2, t2, N);
    agg_init<<<cdiv((long)N * DIN, BS), BS, 0, stream>>>(t2, dinv, bias2, OUT, N, DIN, ostride, ooff);
    agg_edges<<<cdiv((long)E * (DIN / 4), BS), BS, 0, stream>>>(src, dst, ew, dinv, t2, OUT, E, DIN, ostride, ooff);
  };

  gcn_stack(uxb, W1a_f, b1a, W2a_f, b2a, ew1, dinv1, uemb, 2 * DIN, 0);    // u_x_a
  gcn_stack(ixb, W1a_f, b1a, W2a_f, b2a, ew1, dinv1, ixa, DIN, 0);         // i_x_a
  gcn_stack(uxb, W1b_f, b1b, W2b_f, b2b, ew2, dinv2, uemb, 2 * DIN, DIN);  // u_x_a2

  // bf16 copies of MLP gather sources (each row gathered ~E/N times)
  pack_pairs_bf16<<<cdiv((long)N * DIN, BS), BS, 0, stream>>>(uemb, uembb, N * DIN);
  pack_pairs_bf16<<<cdiv((long)N * (DIN / 2), BS), BS, 0, stream>>>(ixa, ixab, N * (DIN / 2));

  // ---- kmeans selection on normal nodes ----
  build_nodeof<<<cdiv(N, BS), BS, 0, stream>>>(atk, A, node_of, N);
  compute_xx<<<cdiv(M, BS), BS, 0, stream>>>(uemb, node_of, xx, M);
  init_cent<<<cdiv(NCL * 2 * DIN, BS), BS, 0, stream>>>(uemb, node_of, cent);
  for (int it = 0; it < KMIT; ++it) {
    fill_f32<<<cdiv(NCL * 2 * DIN, BS), BS, 0, stream>>>(csum, 0.0f, NCL * 2 * DIN);
    fill_f32<<<1, NCL, 0, stream>>>(ccnt, 0.0f, NCL);
    kmeans_assign<<<192, BS, 0, stream>>>(uemb, node_of, xx, cent, csum, ccnt, assign, key, M, 0);
    update_cent<<<cdiv(NCL * 2 * DIN, BS), BS, 0, stream>>>(csum, ccnt, cent);
  }
  kmeans_assign<<<192, BS, 0, stream>>>(uemb, node_of, xx, cent, csum, ccnt, assign, key, M, 1);
  topk_thresh<<<NCL, BS, 0, stream>>>(key, assign, M, kth, needEq);
  fill_i32<<<1, NCL, 0, stream>>>(ctr, 0, NCL);
  fill_f32<<<cdiv(N, BS), BS, 0, stream>>>(umask, 0.0f, N);
  topk_select<<<cdiv(M, BS), BS, 0, stream>>>(key, assign, node_of, kth, needEq, ctr, umask, M);

  // ---- fused edge MLP ----
  mlp_edges<<<cdiv(E >> 4, 8), BS, 0, stream>>>(src, dst, uembb, ixab, mW1_f, mb1,
                                                mW2_f, mb2, mW3, mb3, umask, out, E);
}
